// NPChangeTransitionPrior_2087354106561
// MI455X (gfx1250) — matrix operand order repaired and checked by reading, not compiled, not run on hardware
//
#include <hip/hip_runtime.h>

typedef __attribute__((ext_vector_type(16))) __bf16 v16bf;
typedef __attribute__((ext_vector_type(8)))  float  v8f;

#define BB   256
#define TT   512
#define DD   128
#define HH   128
#define FF   385   // H + L*D + 1
#define TM   510   // T - L
#define RES_ELEMS (BB * TM * DD)   // 16,711,680

union AFrag { v16bf v; unsigned u[8]; };

// ---- pack two f32 -> one dword of bf16x2 (round-half-up) --------------------
// ua+0x8000 rounds bit15 into the kept high half; v_perm_b32 merges the two
// high halves in a single VALU op: dst = {ub.b3, ub.b2, ua.b3, ua.b2}.
__device__ __forceinline__ unsigned pk_bf16(float a, float b) {
  unsigned ua = __builtin_bit_cast(unsigned, a) + 0x8000u;
  unsigned ub = __builtin_bit_cast(unsigned, b) + 0x8000u;
#if __has_builtin(__builtin_amdgcn_perm)
  return __builtin_amdgcn_perm(ub, ua, 0x07060302u);
#else
  return (ua >> 16) | (ub & 0xFFFF0000u);
#endif
}

__device__ __forceinline__ AFrag load_a_frag(const float* arow, int kc, int khalf) {
  const float4* p0 = (const float4*)(arow + kc * 32 + (khalf << 4));
  const float4* p1 = (const float4*)(arow + kc * 32 + 8 + (khalf << 4));
  float4 f0 = p0[0], f1 = p0[1];   // j = 0..7   (K = kc*32 + 16*khalf + 0..7)
  float4 f2 = p1[0], f3 = p1[1];   // j = 8..15  (K = kc*32 + 16*khalf + 8..15)
  AFrag A;
  A.u[0] = pk_bf16(f0.x, f0.y); A.u[1] = pk_bf16(f0.z, f0.w);
  A.u[2] = pk_bf16(f1.x, f1.y); A.u[3] = pk_bf16(f1.z, f1.w);
  A.u[4] = pk_bf16(f2.x, f2.y); A.u[5] = pk_bf16(f2.z, f2.w);
  A.u[6] = pk_bf16(f3.x, f3.y); A.u[7] = pk_bf16(f3.z, f3.w);
  return A;
}

// ---- one 16x16 output tile of a (16 x 128) @ (128 x 128)^T layer ------------
__device__ __forceinline__ void layer16x16(const float* Asrc, int Astride,
                                           const float* W, int Wstride,
                                           const float* bias, float slope,
                                           float* dst, int Dstride,
                                           int lane, int nt) {
  const int lrow  = lane & 15;
  const int khalf = lane >> 4;
  const int n     = nt * 16 + lrow;   // B column = lane & 15
  const v8f vzero = {0.f, 0.f, 0.f, 0.f, 0.f, 0.f, 0.f, 0.f};
  v8f acc = vzero;
  const float* arow = Asrc + lrow * Astride;
#pragma unroll
  for (int kc = 0; kc < 4; ++kc) {
    AFrag A = load_a_frag(arow, kc, khalf);
    const float* wr = W + n * Wstride + kc * 32 + (khalf << 4);
    AFrag Bf;
#pragma unroll
    for (int i = 0; i < 8; ++i) Bf.u[i] = pk_bf16(wr[2 * i], wr[2 * i + 1]);
    acc = __builtin_amdgcn_wmma_f32_16x16x32_bf16(
        false, A.v, false, Bf.v, (short)0, acc, false, false);
  }
  float bv = bias ? bias[n] : 0.0f;
#pragma unroll
  for (int r = 0; r < 8; ++r) {   // D: VGPR r -> row r + 8*khalf, col n
    float v = acc[r] + bv;
    v = fmaxf(v, v * slope);
    dst[(r + (khalf << 3)) * Dstride + n] = v;
  }
}

// ---- Kernel 1: MLP chain + emb @ We^T -> embWe (256x128, workspace) ---------
// 16 blocks x 256 threads; wave w owns N-tile w of the 16x128 layer output.
__global__ __launch_bounds__(256) void mlp_embwe_kernel(
    const float* __restrict__ emb0, const float* __restrict__ W1,
    const float* __restrict__ b1, const float* __restrict__ W2,
    const float* __restrict__ b2, const float* __restrict__ W3,
    const float* __restrict__ b3, const float* __restrict__ Wg,
    float* __restrict__ embWe) {
  __shared__ __align__(16) float hA[16 * 128];
  __shared__ __align__(16) float hB[16 * 128];
  const int lane = threadIdx.x & 31;
  const int wave = threadIdx.x >> 5;   // = nt
  const int m0 = blockIdx.x * 16;

  layer16x16(emb0 + m0 * HH, HH, W1, HH, b1, 0.2f, hA, 128, lane, wave);
  __syncthreads();
  layer16x16(hA, 128, W2, HH, b2, 0.2f, hB, 128, lane, wave);
  __syncthreads();
  layer16x16(hB, 128, W3, HH, b3, 1.0f, hA, 128, lane, wave);   // no activation
  __syncthreads();
  // embWe = emb @ We^T,  We[d,h] = Wg[d*F + h], no bias
  layer16x16(hA, 128, Wg, FF, nullptr, 1.0f, embWe + m0 * DD, DD, lane, wave);
}

// ---- Kernel 2: res = yy @ Wy^T + embWe + xx*wx + bg -------------------------
// M = B*Tm = 130560 rows, K = 256, N = 128.
// 510 blocks x 8 waves; each wave owns TWO 16x128 M-tiles (32 rows).
__global__ __launch_bounds__(256) void gemm_main_kernel(
    const float* __restrict__ x, const float* __restrict__ Wg,
    const float* __restrict__ bg, const float* __restrict__ embWe,
    float* __restrict__ out) {
  __shared__ __align__(32) unsigned BldsU[16384];  // 64 KB: Wy^T as bf16 pairs

  const int tid  = threadIdx.x;
  const int lane = tid & 31;
  const int wave = tid >> 5;

  // Pre-swizzle Wy^T (256x128) into per-lane fragment order (pair-packed):
  // pair p covers j = {2p0, 2p0+1}; value = Wg[n*F + H + k], k = kt*32+16*(l>>4)+j
  for (int p = tid; p < 16384; p += 256) {
    int idx = p << 1;              // even j element index
    int j  = idx & 15;
    int l  = (idx >> 4) & 31;
    int nt = (idx >> 9) & 7;
    int kt = idx >> 12;
    int k  = kt * 32 + ((l >> 4) << 4) + j;
    int n  = nt * 16 + (l & 15);
    const float* src = Wg + n * FF + HH + k;
    BldsU[p] = pk_bf16(src[0], src[1]);
  }
  __syncthreads();

  const int m0    = (blockIdx.x * 8 + wave) * 32;   // 32-row tile group
  const int lrow  = lane & 15;
  const int khalf = lane >> 4;

  const float* arow[2];
#pragma unroll
  for (int h = 0; h < 2; ++h) {
    int m = m0 + h * 16 + lrow;    // yy row = contiguous 256-float window of x
    int b = m / TM;
    int t = m - b * TM;
    arow[h] = x + (b * TT + t) * DD;
  }

  const v8f vzero = {0.f, 0.f, 0.f, 0.f, 0.f, 0.f, 0.f, 0.f};
  v8f acc[2][8];
#pragma unroll
  for (int h = 0; h < 2; ++h)
#pragma unroll
    for (int nt = 0; nt < 8; ++nt) acc[h][nt] = vzero;

#pragma unroll
  for (int kc = 0; kc < 8; ++kc) {
    AFrag A0 = load_a_frag(arow[0], kc, khalf);
    AFrag A1 = load_a_frag(arow[1], kc, khalf);
#pragma unroll
    for (int nt = 0; nt < 8; ++nt) {
      v16bf bf = *(const v16bf*)(BldsU + (((kc * 8 + nt) * 32 + lane) << 3));
      acc[0][nt] = __builtin_amdgcn_wmma_f32_16x16x32_bf16(
          false, A0.v, false, bf, (short)0, acc[0][nt], false, false);
      acc[1][nt] = __builtin_amdgcn_wmma_f32_16x16x32_bf16(
          false, A1.v, false, bf, (short)0, acc[1][nt], false, false);
    }
  }

  // Fused epilogue: + embWe[b,n] + x[b, t+2, n]*wx[n] + bg[n]
  int ebase[2][8], xbase[2][8], obase[2][8];
#pragma unroll
  for (int h = 0; h < 2; ++h) {
#pragma unroll
    for (int r = 0; r < 8; ++r) {
      int row = m0 + h * 16 + r + (khalf << 3);
      int bb  = row / TM;
      int tt  = row - bb * TM;
      ebase[h][r] = bb * DD;
      xbase[h][r] = (bb * TT + tt + 2) * DD;
      obase[h][r] = row * DD;
    }
  }
#pragma unroll
  for (int nt = 0; nt < 8; ++nt) {
    int n = nt * 16 + lrow;
    float wxn = Wg[n * FF + (FF - 1)];
    float bgn = bg[n];
#pragma unroll
    for (int h = 0; h < 2; ++h) {
#pragma unroll
      for (int r = 0; r < 8; ++r) {
        float v = acc[h][nt][r] + embWe[ebase[h][r] + n]
                + x[xbase[h][r] + n] * wxn + bgn;
        out[obase[h][r] + n] = v;
      }
    }
  }
}

// ---- Kernel 3: sum_log_abs_det = Tm * sum_d log|wx[d]|, broadcast to B ------
__global__ __launch_bounds__(256) void logdet_kernel(const float* __restrict__ Wg,
                                                     float* __restrict__ out) {
  float s = 0.0f;
  for (int d = 0; d < DD; ++d) s += logf(fabsf(Wg[d * FF + (FF - 1)]));
  out[RES_ELEMS + threadIdx.x] = s * (float)TM;
}

extern "C" void kernel_launch(void* const* d_in, const int* in_sizes, int n_in,
                              void* d_out, int out_size, void* d_ws, size_t ws_size,
                              hipStream_t stream) {
  (void)in_sizes; (void)n_in; (void)out_size; (void)ws_size;
  const float* x    = (const float*)d_in[0];
  const float* emb0 = (const float*)d_in[1];
  const float* W1   = (const float*)d_in[2];
  const float* b1   = (const float*)d_in[3];
  const float* W2   = (const float*)d_in[4];
  const float* b2   = (const float*)d_in[5];
  const float* W3   = (const float*)d_in[6];
  const float* b3   = (const float*)d_in[7];
  const float* Wg   = (const float*)d_in[8];
  const float* bg   = (const float*)d_in[9];
  float* out   = (float*)d_out;
  float* embWe = (float*)d_ws;   // 256*128 f32 = 128 KB scratch

  mlp_embwe_kernel<<<16, 256, 0, stream>>>(emb0, W1, b1, W2, b2, W3, b3, Wg, embWe);
  gemm_main_kernel<<<(BB * TM) / 256, 256, 0, stream>>>(x, Wg, bg, embWe, out);
  logdet_kernel<<<1, 256, 0, stream>>>(Wg, out);
}
